// LunarisCodex_12601434047135
// MI455X (gfx1250) — matrix operand order repaired and checked by reading, not compiled, not run on hardware
//
#include <hip/hip_runtime.h>
#include <hip/hip_bf16.h>
#include <math.h>

// ---------------------------------------------------------------------------
// Model constants (from reference)
// ---------------------------------------------------------------------------
#define Bc    2
#define Tc    1024
#define Dc    768
#define NHc   12
#define HDc   64
#define Vc    50257
#define Lc    2
#define Ec    8
#define Hc    2048
#define Nc    (Bc*Tc)          // 2048
#define CAPc  320              // ceil(N/E*1.25)
#define AUXW  0.01f
#define ZW    0.001f
#define EPSc  1e-5f

typedef __attribute__((ext_vector_type(16))) __bf16        v16bf;
typedef __attribute__((ext_vector_type(8)))  float         v8f;
typedef __attribute__((ext_vector_type(4)))  unsigned int  u32x4;
typedef __attribute__((ext_vector_type(8)))  int           i32x8;
typedef __attribute__((ext_vector_type(4)))  int           i32x4;

// float -> bf16 bits, round-to-nearest-even
__device__ __forceinline__ unsigned short f2bfb(float f) {
    unsigned int u = __float_as_uint(f);
    u += 0x7FFFu + ((u >> 16) & 1u);
    return (unsigned short)(u >> 16);
}
__device__ __forceinline__ __bf16 us2bf(unsigned short s) {
    return __builtin_bit_cast(__bf16, s);
}

// LDS byte offset of a __shared__ object (generic AS: low 32 bits = LDS offset)
__device__ __forceinline__ unsigned lds_off(const void* p) {
    return (unsigned)(unsigned long long)(uintptr_t)p;
}

// ---------------------------------------------------------------------------
// TDM: async 2-D bf16 tile load Global -> LDS with in-flight row padding.
// Tile: tileD1 rows x tileD0 elements (dim0 contiguous, 2-byte elements).
// Rows of 32 elems = 64 B; pad_interval=16 DWORDs, pad_amount=1 DWORD
// => LDS row pitch 68 B, matching ushort[64][34] tiles.
// Tensor dims measured from the tile start: OOB rows/cols read as zero.
// ---------------------------------------------------------------------------
__device__ __forceinline__ void tdm_load_2d(unsigned ldsOff, const void* gp,
                                            unsigned tileD0, unsigned tileD1,
                                            unsigned tensD0, unsigned tensD1,
                                            unsigned long long strideD0 /*elems*/)
{
    unsigned long long ga = (unsigned long long)(uintptr_t)gp;
    u32x4 g0;
    g0[0] = 1u;                                              // count=1, user D#
    g0[1] = ldsOff;                                          // lds_addr (bytes)
    g0[2] = (unsigned)ga;                                    // global_addr[31:0]
    g0[3] = (unsigned)((ga >> 32) & 0x01FFFFFFu) | (2u << 30); // addr[56:32]|type=2
    i32x8 g1;
    g1[0] = (int)((1u << 16)      // data_size = 1 (2 bytes)
                | (1u << 20)      // pad_enable
                | (3u << 22));    // pad_interval = 16 DWORDs; pad_amount=1 DWORD
    g1[1] = (int)((tensD0 & 0xFFFFu) << 16);                 // tensor_dim0[15:0]
    g1[2] = (int)(((tensD0 >> 16) & 0xFFFFu) |
                  ((tensD1 & 0xFFFFu) << 16));               // td0[31:16]|td1[15:0]
    g1[3] = (int)(((tensD1 >> 16) & 0xFFFFu) |
                  ((tileD0 & 0xFFFFu) << 16));               // td1[31:16]|tile_dim0
    g1[4] = (int)(tileD1 & 0xFFFFu);                         // tile_dim1 (dim2=0)
    g1[5] = (int)(unsigned)(strideD0 & 0xFFFFFFFFu);         // dim0_stride[31:0]
    g1[6] = (int)(unsigned)((strideD0 >> 32) & 0xFFFFu);     // dim0_stride[47:32]
    g1[7] = 0;
    i32x4 gz4 = {0, 0, 0, 0};
    i32x8 gz8 = {0, 0, 0, 0, 0, 0, 0, 0};
    __builtin_amdgcn_tensor_load_to_lds(g0, g1, gz4, gz4, gz8, 0);
}

// ---------------------------------------------------------------------------
// WMMA GEMM, TDM-fed, double buffered.
//   A : bf16 [M x K]   row-major, lda = row pitch (elems)
//   Bt: bf16 [N x K]   row-major (pre-transposed weights), ldb = pitch
//   C : f32  [M x N]   (+ optional f32 residual)
// Block: 256 threads (8 wave32). Tile 64x64, K-step 32.
// Wave w owns rows [(w&3)*16,+16), cols [(w>>2)*32,+32) of the tile.
// ---------------------------------------------------------------------------
__global__ __launch_bounds__(256)
void gemm_wmma(const unsigned short* __restrict__ A, long lda, long sA,
               const unsigned short* __restrict__ Bt, long ldb, long sB,
               float* __restrict__ C, long ldc, long sC,
               const float* __restrict__ Res, long ldr, long sR,
               int M, int Ncol, int K)
{
    __shared__ unsigned short As[2][64][34];
    __shared__ unsigned short Bs[2][64][34];

    int z = blockIdx.z;
    A  += (long)z * sA;
    Bt += (long)z * sB;
    C  += (long)z * sC;
    if (Res) Res += (long)z * sR;

    const int row0 = blockIdx.y * 64;
    const int col0 = blockIdx.x * 64;
    const int tid  = threadIdx.x;
    const int lane = tid & 31;
    const int wid  = tid >> 5;
    const int wm   = wid & 3;
    const int wn2  = wid >> 2;
    const int hlf  = lane >> 4;
    const int lnn  = lane & 15;
    const int nk   = K >> 5;

    v8f acc[2] = {};

    auto issue = [&](int s) {
        if (s >= nk) return;
        int k0  = s << 5;
        int buf = s & 1;
        tdm_load_2d(lds_off(&As[buf][0][0]),
                    A + (long)row0 * lda + k0,
                    32, 64,
                    (unsigned)(K - k0), (unsigned)(M - row0),
                    (unsigned long long)lda);
        tdm_load_2d(lds_off(&Bs[buf][0][0]),
                    Bt + (long)col0 * ldb + k0,
                    32, 64,
                    (unsigned)(K - k0), (unsigned)(Ncol - col0),
                    (unsigned long long)ldb);
    };

    if (wid == 0) { issue(0); issue(1); }

    for (int s = 0; s < nk; ++s) {
        if (wid == 0) {
            if (s + 1 < nk) __builtin_amdgcn_s_wait_tensorcnt(2);
            else            __builtin_amdgcn_s_wait_tensorcnt(0);
        }
        __syncthreads();

        const unsigned short (*Ab)[34] = As[s & 1];
        const unsigned short (*Bb)[34] = Bs[s & 1];

        v16bf a;
        int am = wm * 16 + lnn;
        #pragma unroll
        for (int i = 0; i < 16; ++i) {
            int kk = (i < 8 ? i : i + 8) + 8 * hlf;
            a[i] = us2bf(Ab[am][kk]);
        }
        #pragma unroll
        for (int c = 0; c < 2; ++c) {
            v16bf bv;
            int bn = wn2 * 32 + c * 16 + lnn;
            #pragma unroll
            for (int i = 0; i < 16; ++i)
                bv[i] = us2bf(Bb[bn][(i < 8 ? i : i + 8) + 8 * hlf]);
            acc[c] = __builtin_amdgcn_wmma_f32_16x16x32_bf16(
                false, a, false, bv, (short)0, acc[c], false, false);
        }
        __syncthreads();
        if (wid == 0) issue(s + 2);
    }

    #pragma unroll
    for (int c = 0; c < 2; ++c)
        #pragma unroll
        for (int r = 0; r < 8; ++r) {
            int gr = row0 + wm * 16 + r + 8 * hlf;
            int gc = col0 + wn2 * 32 + c * 16 + lnn;
            if (gr < M && gc < Ncol) {
                float v = acc[c][r];
                if (Res) v += Res[(long)gr * ldr + gc];
                C[(long)gr * ldc + gc] = v;
            }
        }
}

// ---------------------------------------------------------------------------
// fp32 -> bf16 elementwise convert
// ---------------------------------------------------------------------------
__global__ void cvt_kernel(const float* __restrict__ in,
                           unsigned short* __restrict__ outp, long count)
{
    long i = (long)blockIdx.x * 256 + threadIdx.x;
    if (i < count) outp[i] = f2bfb(in[i]);
}

// qkv fp32 -> bf16, folding 1/sqrt(HD) into q columns
__global__ void cvt_qkv_kernel(const float* __restrict__ in,
                               unsigned short* __restrict__ outp)
{
    long i = (long)blockIdx.x * 256 + threadIdx.x;
    if (i >= (long)Nc * 3 * Dc) return;
    int col = (int)(i % (3 * Dc));
    float v = in[i];
    if (col < Dc) v *= 0.125f;
    outp[i] = f2bfb(v);
}

// ---------------------------------------------------------------------------
// Tiled transpose + convert: W fp32 [K x Ncol] -> Wt bf16 [Ncol x K]
// ---------------------------------------------------------------------------
__global__ void transcvt_kernel(const float* __restrict__ in,
                                unsigned short* __restrict__ outp,
                                int K, int Ncol, long sIn, long sOut)
{
    __shared__ float tile[32][33];
    long z = blockIdx.z;
    const float* I = in + z * sIn;
    unsigned short* O = outp + z * sOut;
    int kb = blockIdx.y * 32, nb = blockIdx.x * 32;
    for (int j = threadIdx.y; j < 32; j += 8) {
        int k = kb + j, n = nb + threadIdx.x;
        tile[j][threadIdx.x] = (k < K && n < Ncol) ? I[(long)k * Ncol + n] : 0.f;
    }
    __syncthreads();
    for (int j = threadIdx.y; j < 32; j += 8) {
        int n = nb + j, k = kb + threadIdx.x;
        if (n < Ncol && k < K)
            O[(long)n * K + k] = f2bfb(tile[threadIdx.x][j]);
    }
}

// ---------------------------------------------------------------------------
// Embedding gather: x[n,d] = wte[idx[n], d]
// ---------------------------------------------------------------------------
__global__ void embed_kernel(const int* __restrict__ idx,
                             const float* __restrict__ wte,
                             float* __restrict__ x)
{
    long i = (long)blockIdx.x * 256 + threadIdx.x;
    if (i < (long)Nc * Dc) {
        int n = (int)(i / Dc), d = (int)(i % Dc);
        x[i] = wte[(long)idx[n] * Dc + d];
    }
}

// ---------------------------------------------------------------------------
// QK rmsnorm (over full D, matching reference) + per-head RoPE, in place (fp32)
// ---------------------------------------------------------------------------
__global__ void qk_prep_kernel(float* __restrict__ qkv,
                               const float* __restrict__ qw,
                               const float* __restrict__ kw)
{
    int n = blockIdx.x, tid = threadIdx.x;
    int pos = n % Tc;
    __shared__ float red[256];
    float* base = qkv + (long)n * (3 * Dc);
    for (int part = 0; part < 2; ++part) {
        float* p = base + part * Dc;
        const float* w = part ? kw : qw;
        float ss = 0.f;
        for (int d = tid; d < Dc; d += 256) { float v = p[d]; ss += v * v; }
        red[tid] = ss; __syncthreads();
        for (int s = 128; s > 0; s >>= 1) {
            if (tid < s) red[tid] += red[tid + s];
            __syncthreads();
        }
        float rms = rsqrtf(red[0] / (float)Dc + EPSc);
        __syncthreads();
        for (int pidx = tid; pidx < (Dc / 2); pidx += 256) {
            int hh = pidx / (HDc / 2), i = pidx % (HDc / 2);
            int d0 = hh * HDc + 2 * i;
            float x0 = p[d0]     * rms * w[d0];
            float x1 = p[d0 + 1] * rms * w[d0 + 1];
            float fr = __expf(-((float)(2 * i) / (float)HDc) * 9.210340371976184f);
            float ang = (float)pos * fr;
            float c = __cosf(ang), s = __sinf(ang);
            p[d0]     = x0 * c - x1 * s;
            p[d0 + 1] = x0 * s + x1 * c;
        }
        __syncthreads();
    }
}

// ---------------------------------------------------------------------------
// Generic rmsnorm over rows of length Dc
// ---------------------------------------------------------------------------
__global__ void rmsnorm_kernel(const float* __restrict__ x,
                               const float* __restrict__ w,
                               float* __restrict__ y)
{
    int n = blockIdx.x, tid = threadIdx.x;
    __shared__ float red[256];
    const float* xr = x + (long)n * Dc;
    float ss = 0.f;
    for (int d = tid; d < Dc; d += 256) { float v = xr[d]; ss += v * v; }
    red[tid] = ss; __syncthreads();
    for (int s = 128; s > 0; s >>= 1) {
        if (tid < s) red[tid] += red[tid + s];
        __syncthreads();
    }
    float rms = rsqrtf(red[0] / (float)Dc + EPSc);
    for (int d = tid; d < Dc; d += 256)
        y[(long)n * Dc + d] = xr[d] * rms * w[d];
}

// ---------------------------------------------------------------------------
// Flash attention, one wave per (b, h, 16-query tile). bf16 qkv input,
// q pre-scaled. QK^T and P*V via wmma_f32_16x16x32_bf16, online softmax.
// ---------------------------------------------------------------------------
__global__ __launch_bounds__(32)
void attn_kernel(const unsigned short* __restrict__ qkvb, float* __restrict__ y)
{
    const int lane = threadIdx.x;
    const int qt = blockIdx.x, h = blockIdx.y, b = blockIdx.z;
    const int hlf = lane >> 4, lnn = lane & 15;

    v16bf qa[2];
    {
        int qrow = qt * 16 + lnn;
        long baseq = (long)(b * Tc + qrow) * (3 * Dc) + h * HDc;
        #pragma unroll
        for (int j = 0; j < 2; ++j)
            #pragma unroll
            for (int i = 0; i < 16; ++i) {
                int k = (i < 8 ? i : i + 8) + 8 * hlf;
                qa[j][i] = us2bf(qkvb[baseq + j * 32 + k]);
            }
    }

    v8f o[4] = {};
    float mrow[8], lsum[8];
    #pragma unroll
    for (int r = 0; r < 8; ++r) { mrow[r] = -3.0e38f; lsum[r] = 0.f; }

    __shared__ unsigned short Ps[16][34];

    const int qmax = qt * 16 + 15;
    const int ktiles = qmax / 32 + 1;

    for (int kt = 0; kt < ktiles; ++kt) {
        int kb = kt * 32;
        v8f s[2] = {};
        #pragma unroll
        for (int c = 0; c < 2; ++c) {
            int key = kb + c * 16 + lnn;
            long basek = (long)(b * Tc + key) * (3 * Dc) + Dc + h * HDc;
            #pragma unroll
            for (int j = 0; j < 2; ++j) {
                v16bf kf;
                #pragma unroll
                for (int i = 0; i < 16; ++i) {
                    int hd = j * 32 + (i < 8 ? i : i + 8) + 8 * hlf;
                    kf[i] = us2bf(qkvb[basek + hd]);
                }
                s[c] = __builtin_amdgcn_wmma_f32_16x16x32_bf16(
                    false, qa[j], false, kf, (short)0, s[c], false, false);
            }
        }
        float pr0[8], pr1[8];
        #pragma unroll
        for (int r = 0; r < 8; ++r) {
            int q = qt * 16 + r + 8 * hlf;
            float s0 = (kb + lnn      <= q) ? s[0][r] : -1.0e30f;
            float s1 = (kb + 16 + lnn <= q) ? s[1][r] : -1.0e30f;
            float tmax = fmaxf(s0, s1);
            #pragma unroll
            for (int m = 1; m < 16; m <<= 1)
                tmax = fmaxf(tmax, __shfl_xor(tmax, m, 32));
            float nm = fmaxf(mrow[r], tmax);
            float corr = __expf(mrow[r] - nm);
            float p0 = __expf(s0 - nm);
            float p1 = __expf(s1 - nm);
            float rsum = p0 + p1;
            #pragma unroll
            for (int m = 1; m < 16; m <<= 1)
                rsum += __shfl_xor(rsum, m, 32);
            lsum[r] = lsum[r] * corr + rsum;
            mrow[r] = nm;
            #pragma unroll
            for (int c2 = 0; c2 < 4; ++c2) o[c2][r] *= corr;
            pr0[r] = p0; pr1[r] = p1;
        }
        #pragma unroll
        for (int r = 0; r < 8; ++r) {
            Ps[r + 8 * hlf][lnn]      = f2bfb(pr0[r]);
            Ps[r + 8 * hlf][16 + lnn] = f2bfb(pr1[r]);
        }
        __syncthreads();
        v16bf pa;
        #pragma unroll
        for (int i = 0; i < 16; ++i) {
            int k = (i < 8 ? i : i + 8) + 8 * hlf;
            pa[i] = us2bf(Ps[lnn][k]);
        }
        #pragma unroll
        for (int c2 = 0; c2 < 4; ++c2) {
            v16bf vf;
            int hd = c2 * 16 + lnn;
            #pragma unroll
            for (int i = 0; i < 16; ++i) {
                int key = kb + (i < 8 ? i : i + 8) + 8 * hlf;
                vf[i] = us2bf(qkvb[(long)(b * Tc + key) * (3 * Dc) + 2 * Dc + h * HDc + hd]);
            }
            o[c2] = __builtin_amdgcn_wmma_f32_16x16x32_bf16(
                false, pa, false, vf, (short)0, o[c2], false, false);
        }
        __syncthreads();
    }

    #pragma unroll
    for (int c2 = 0; c2 < 4; ++c2)
        #pragma unroll
        for (int r = 0; r < 8; ++r) {
            int q = qt * 16 + r + 8 * hlf;
            int hd = c2 * 16 + lnn;
            y[(long)(b * Tc + q) * Dc + h * HDc + hd] = o[c2][r] / lsum[r];
        }
}

// ---------------------------------------------------------------------------
// MoE gate logits: glog[n,e] = sum_d ffn_in[n,d] * gate_w[d,e]
// ---------------------------------------------------------------------------
__global__ void gate_kernel(const float* __restrict__ x,
                            const float* __restrict__ gw,
                            float* __restrict__ glog)
{
    int n = blockIdx.x, tid = threadIdx.x;
    __shared__ float red[256];
    int e = tid & 7, chunk = tid >> 3;
    const float* xr = x + (long)n * Dc;
    float s = 0.f;
    int d0 = chunk * 24;
    for (int d = d0; d < d0 + 24; ++d) s += xr[d] * gw[d * Ec + e];
    red[tid] = s; __syncthreads();
    if (tid < 8) {
        float t = 0.f;
        for (int j = 0; j < 32; ++j) t += red[j * 8 + tid];
        glog[(long)n * Ec + tid] = t;
    }
}

__global__ void router_kernel(const float* __restrict__ glog,
                              int* __restrict__ top,
                              float* __restrict__ pm,
                              float* __restrict__ cnt,
                              float* __restrict__ zsq)
{
    int n = blockIdx.x * 256 + threadIdx.x;
    if (n >= Nc) return;
    const float* l = glog + (long)n * Ec;
    float m = l[0]; int am = 0;
    #pragma unroll
    for (int e = 1; e < Ec; ++e) if (l[e] > m) { m = l[e]; am = e; }
    float ex[Ec], s = 0.f;
    #pragma unroll
    for (int e = 0; e < Ec; ++e) { ex[e] = __expf(l[e] - m); s += ex[e]; }
    float inv = 1.f / s;
    #pragma unroll
    for (int e = 0; e < Ec; ++e) atomicAdd(&pm[e], ex[e] * inv);
    atomicAdd(&cnt[am], 1.0f);
    float z = __logf(s) + m;
    atomicAdd(zsq, z * z);
    top[n] = am;
}

// deterministic stable ranking in token order (matches stable argsort)
__global__ void rank_kernel(const int* __restrict__ top,
                            int* __restrict__ rank,
                            int* __restrict__ keep)
{
    if (blockIdx.x == 0 && threadIdx.x == 0) {
        int c[Ec];
        for (int e = 0; e < Ec; ++e) c[e] = 0;
        for (int n = 0; n < Nc; ++n) {
            int e = top[n];
            int r = c[e]++;
            rank[n] = r;
            keep[n] = (r < CAPc) ? 1 : 0;
        }
    }
}

__global__ void aux_kernel(const float* __restrict__ pm,
                           const float* __restrict__ cnt,
                           const float* __restrict__ zsq,
                           float* __restrict__ aux_acc)
{
    if (blockIdx.x == 0 && threadIdx.x == 0) {
        float bal = 0.f;
        for (int e = 0; e < Ec; ++e)
            bal += (pm[e] / (float)Nc) * (cnt[e] / (float)Nc);
        *aux_acc += bal * AUXW * (float)Ec + (zsq[0] / (float)Nc) * ZW;
    }
}

// gather kept tokens into expert buffers, converting to bf16
__global__ void gather_kernel(const float* __restrict__ ffn_in,
                              const int* __restrict__ top,
                              const int* __restrict__ rank,
                              const int* __restrict__ keep,
                              unsigned short* __restrict__ xbb)
{
    long i = (long)blockIdx.x * 256 + threadIdx.x;
    if (i >= (long)Nc * Dc) return;
    int n = (int)(i / Dc), d = (int)(i % Dc);
    if (keep[n])
        xbb[((long)top[n] * CAPc + rank[n]) * Dc + d] = f2bfb(ffn_in[i]);
}

// silu(g)*u, written directly as bf16 for the w2 GEMM
__global__ void silu_kernel(const float* __restrict__ gu,
                            unsigned short* __restrict__ h1b)
{
    long i = (long)blockIdx.x * 256 + threadIdx.x;
    if (i >= (long)Ec * CAPc * Hc) return;
    long ec = i / Hc; int hh = (int)(i % Hc);
    float g = gu[ec * (2 * Hc) + hh];
    float u = gu[ec * (2 * Hc) + Hc + hh];
    h1b[i] = f2bfb(g / (1.f + __expf(-g)) * u);
}

__global__ void combine_kernel(const float* __restrict__ h,
                               const float* __restrict__ yexp,
                               const int* __restrict__ top,
                               const int* __restrict__ rank,
                               const int* __restrict__ keep,
                               float* __restrict__ x)
{
    long i = (long)blockIdx.x * 256 + threadIdx.x;
    if (i >= (long)Nc * Dc) return;
    int n = (int)(i / Dc), d = (int)(i % Dc);
    float v = h[i];
    if (keep[n])
        v += yexp[((long)top[n] * CAPc + rank[n]) * Dc + d];
    x[i] = v;
}

__global__ void finalize_kernel(const float* __restrict__ aux_acc,
                                float* __restrict__ out)
{
    if (blockIdx.x == 0 && threadIdx.x == 0)
        out[(long)Nc * Vc] = *aux_acc;
}

// ---------------------------------------------------------------------------
// Host side
// ---------------------------------------------------------------------------
static inline void launch_gemm(hipStream_t st,
                               const unsigned short* A, long lda, long sA,
                               const unsigned short* Bt, long ldb, long sB,
                               float* C, long ldc, long sC,
                               const float* R, long ldr, long sR,
                               int M, int Ncol, int K, int batch)
{
    dim3 g((Ncol + 63) / 64, (M + 63) / 64, batch);
    gemm_wmma<<<g, 256, 0, st>>>(A, lda, sA, Bt, ldb, sB, C, ldc, sC,
                                 R, ldr, sR, M, Ncol, K);
}

static inline void launch_transcvt(hipStream_t st, const float* in, unsigned short* outp,
                                   int K, int Ncol, long sIn, long sOut, int batch)
{
    dim3 g((Ncol + 31) / 32, (K + 31) / 32, batch);
    transcvt_kernel<<<g, dim3(32, 8), 0, st>>>(in, outp, K, Ncol, sIn, sOut);
}

static inline void launch_cvt(hipStream_t st, const float* in, unsigned short* outp, long count)
{
    cvt_kernel<<<(int)((count + 255) / 256), 256, 0, st>>>(in, outp, count);
}

extern "C" void kernel_launch(void* const* d_in, const int* in_sizes, int n_in,
                              void* d_out, int out_size, void* d_ws, size_t ws_size,
                              hipStream_t stream)
{
    const int*   idx   = (const int*)  d_in[0];
    const float* wte   = (const float*)d_in[1];
    const float* wqkv  = (const float*)d_in[2];
    const float* qnw   = (const float*)d_in[3];
    const float* knw   = (const float*)d_in[4];
    const float* oproj = (const float*)d_in[5];
    const float* ffnw  = (const float*)d_in[6];
    const float* gatew = (const float*)d_in[7];
    const float* w13   = (const float*)d_in[8];
    const float* w2    = (const float*)d_in[9];
    const float* lnf   = (const float*)d_in[10];
    float* out = (float*)d_out;
    float* w   = (float*)d_ws;

    // ---- fp32 region (floats) ----
    const long F_QKV = 0;                                  // N*3D
    const long F_X   = F_QKV + (long)Nc * 3 * Dc;          // N*D
    const long F_AY  = F_X   + (long)Nc * Dc;              // N*D
    const long F_H   = F_AY  + (long)Nc * Dc;              // N*D
    const long F_FFN = F_H   + (long)Nc * Dc;              // N*D
    const long F_GU  = F_FFN + (long)Nc * Dc;              // E*CAP*2H
    const long F_YE  = F_GU  + (long)Ec * CAPc * 2 * Hc;   // E*CAP*D
    const long F_GL  = F_YE  + (long)Ec * CAPc * Dc;       // N*E
    const long F_RED = F_GL  + (long)Nc * Ec;              // 32
    const long F_END = F_RED + 32;

    float* qkv  = w + F_QKV;
    float* x    = w + F_X;
    float* ay   = w + F_AY;
    float* hbuf = w + F_H;
    float* ffn  = w + F_FFN;
    float* gu   = w + F_GU;
    float* ye   = w + F_YE;
    float* glog = w + F_GL;
    float* pm   = w + F_RED;
    float* cnt  = pm + 8;
    float* zsq  = pm + 16;
    float* aux  = pm + 17;

    // ---- int region ----
    int* top   = (int*)(w + F_END);
    int* rankb = top + Nc;
    int* keepb = top + 2 * Nc;

    // ---- bf16 region (ushorts) ----
    unsigned short* us = (unsigned short*)(top + 3 * Nc);
    const long U_QKVB = 0;                                 // N*3D
    const long U_ABF  = U_QKVB + (long)Nc * 3 * Dc;        // N*D  (activation A)
    const long U_XBB  = U_ABF  + (long)Nc * Dc;            // E*CAP*D
    const long U_H1B  = U_XBB  + (long)Ec * CAPc * Dc;     // E*CAP*H
    const long U_WTB  = U_H1B  + (long)Ec * CAPc * Hc;     // max(V*D, E*D*2H)
    unsigned short* qkvb = us + U_QKVB;
    unsigned short* abf  = us + U_ABF;
    unsigned short* xbb  = us + U_XBB;
    unsigned short* h1b  = us + U_H1B;
    unsigned short* wtb  = us + U_WTB;

    const long ND = (long)Nc * Dc;
    const int  nblkND = (int)((ND + 255) / 256);

    (void)hipMemsetAsync(aux, 0, sizeof(float), stream);

    embed_kernel<<<nblkND, 256, 0, stream>>>(idx, wte, x);

    for (int l = 0; l < Lc; ++l) {
        const float* wqkv_l = wqkv + (long)l * Dc * 3 * Dc;
        const float* op_l   = oproj + (long)l * Dc * Dc;
        const float* gw_l   = gatew + (long)l * Dc * Ec;
        const float* w13_l  = w13 + (long)l * Ec * Dc * 2 * Hc;
        const float* w2_l   = w2  + (long)l * Ec * Hc * Dc;

        // qkv = x @ wqkv[l]
        launch_cvt(stream, x, abf, ND);
        launch_transcvt(stream, wqkv_l, wtb, Dc, 3 * Dc, 0, 0, 1);
        launch_gemm(stream, abf, Dc, 0, wtb, Dc, 0, qkv, 3 * Dc, 0,
                    nullptr, 0, 0, Nc, 3 * Dc, Dc, 1);
        // q/k rmsnorm + rope (fp32 in place), then bf16 with q scale folded in
        qk_prep_kernel<<<Nc, 256, 0, stream>>>(qkv, qnw + (long)l * Dc, knw + (long)l * Dc);
        cvt_qkv_kernel<<<(int)(((long)Nc * 3 * Dc + 255) / 256), 256, 0, stream>>>(qkv, qkvb);
        // flash attention
        attn_kernel<<<dim3(Tc / 16, NHc, Bc), 32, 0, stream>>>(qkvb, ay);
        // h = ay @ o_proj[l] + x
        launch_cvt(stream, ay, abf, ND);
        launch_transcvt(stream, op_l, wtb, Dc, Dc, 0, 0, 1);
        launch_gemm(stream, abf, Dc, 0, wtb, Dc, 0, hbuf, Dc, 0,
                    x, Dc, 0, Nc, Dc, Dc, 1);
        // ffn_in = rmsnorm(h)
        rmsnorm_kernel<<<Nc, 256, 0, stream>>>(hbuf, ffnw + (long)l * Dc, ffn);

        // MoE routing
        (void)hipMemsetAsync(pm, 0, 17 * sizeof(float), stream);
        gate_kernel<<<Nc, 256, 0, stream>>>(ffn, gw_l, glog);
        router_kernel<<<(Nc + 255) / 256, 256, 0, stream>>>(glog, top, pm, cnt, zsq);
        rank_kernel<<<1, 1, 0, stream>>>(top, rankb, keepb);
        aux_kernel<<<1, 1, 0, stream>>>(pm, cnt, zsq, aux);

        (void)hipMemsetAsync(xbb, 0, (size_t)Ec * CAPc * Dc * sizeof(unsigned short), stream);
        gather_kernel<<<nblkND, 256, 0, stream>>>(ffn, top, rankb, keepb, xbb);

        // gu[e] = xb[e] @ w13[l,e]
        launch_transcvt(stream, w13_l, wtb, Dc, 2 * Hc,
                        (long)Dc * 2 * Hc, (long)2 * Hc * Dc, Ec);
        launch_gemm(stream, xbb, Dc, (long)CAPc * Dc,
                    wtb, Dc, (long)2 * Hc * Dc,
                    gu, 2 * Hc, (long)CAPc * 2 * Hc,
                    nullptr, 0, 0, CAPc, 2 * Hc, Dc, Ec);
        // h1 = silu(g)*u  (bf16)
        {
            long tot = (long)Ec * CAPc * Hc;
            silu_kernel<<<(int)((tot + 255) / 256), 256, 0, stream>>>(gu, h1b);
        }
        // ye[e] = h1[e] @ w2[l,e]
        launch_transcvt(stream, w2_l, wtb, Hc, Dc,
                        (long)Hc * Dc, (long)Dc * Hc, Ec);
        launch_gemm(stream, h1b, Hc, (long)CAPc * Hc,
                    wtb, Hc, (long)Dc * Hc,
                    ye, Dc, (long)CAPc * Dc,
                    nullptr, 0, 0, CAPc, Dc, Hc, Ec);
        // x = h + scatter(ye)
        combine_kernel<<<nblkND, 256, 0, stream>>>(hbuf, ye, top, rankb, keepb, x);
    }

    // final norm + logits = xn @ wte^T   (wte is already [V][K]; convert only)
    rmsnorm_kernel<<<Nc, 256, 0, stream>>>(x, lnf, ffn);
    launch_cvt(stream, ffn, abf, ND);
    launch_cvt(stream, wte, wtb, (long)Vc * Dc);
    launch_gemm(stream, abf, Dc, 0, wtb, Dc, 0, out, Vc, 0,
                nullptr, 0, 0, Nc, Vc, Dc, 1);

    finalize_kernel<<<1, 1, 0, stream>>>(aux, out);
}
